// MsaHmmCell_25168508354912
// MI455X (gfx1250) — matrix operand order repaired and checked by reading, not compiled, not run on hardware
//
#include <hip/hip_runtime.h>

// HMM forward (scaled): T=512, B=128, Q=1027, D=26.
// Persistent kernel; sf @ A via bf16x3 error-compensated WMMA
// (v_wmma_f32_16x16x32_bf16, fp32 accumulate).

#define TT   512
#define BB   128
#define QQ   1027
#define QP   1040          // column padding (65 * 16)
#define NKC  33            // K chunks of 32
#define SFP  1056          // K-padded sf row length (NKC*32)
#define DD   26
#define NCT  65
#define NRT  8
#define NBLK 64
#define NTHR 256
#define NT   (NBLK * NTHR)
#define NWAVES (NT / 32)

typedef __attribute__((ext_vector_type(16))) __bf16 v16bf;
typedef __attribute__((ext_vector_type(8)))  __bf16 v8bf;
typedef __attribute__((ext_vector_type(8)))  float  v8f;

__device__ __forceinline__ void grid_sync(unsigned* cnt, unsigned& phase) {
  __syncthreads();
  if (threadIdx.x == 0) {
    __threadfence();
    atomicAdd(cnt, 1u);
    const unsigned target = (phase + 1u) * (unsigned)NBLK;
    while (__hip_atomic_load(cnt, __ATOMIC_ACQUIRE, __HIP_MEMORY_SCOPE_AGENT) < target) {
      __builtin_amdgcn_s_sleep(2);
    }
    __threadfence();
  }
  __syncthreads();
  ++phase;
}

__device__ __forceinline__ float edot(const float* __restrict__ xr,
                                      const float* __restrict__ br) {
  float e = 0.f;
#pragma unroll
  for (int d = 0; d < DD; ++d) e = fmaf(xr[d], br[d], e);
  return e;
}

// Build a v16bf A-fragment from two contiguous 16-byte chunks.
__device__ __forceinline__ v16bf ld16(const __bf16* p0, const __bf16* p1) {
  v8bf a = *(const v8bf*)p0;
  v8bf b = *(const v8bf*)p1;
  return __builtin_shufflevector(a, b, 0, 1, 2, 3, 4, 5, 6, 7,
                                 8, 9, 10, 11, 12, 13, 14, 15);
}

// Phase B: U[b,q] = Et[b,q] * sum_k sf[b,k]*A[k,q]; Sp[b,ct] = tile row sums.
__device__ void phase_gemm(const __bf16* __restrict__ Apk_hi,
                           const __bf16* __restrict__ Apk_lo,
                           const __bf16* __restrict__ sf_hi,
                           const __bf16* __restrict__ sf_lo,
                           const float* __restrict__ Et,
                           float* __restrict__ U, float* __restrict__ Sp,
                           unsigned wave, unsigned lane) {
  const unsigned half   = lane >> 4;           // 0: lanes 0-15, 1: lanes 16-31
  const unsigned lane15 = lane & 15u;
  for (unsigned tile = wave; tile < NRT * NCT; tile += NWAVES) {
    const unsigned rt = tile / NCT, ct = tile % NCT;
    const unsigned m0 = rt * 16u, n0 = ct * 16u;
    const unsigned mA = m0 + lane15;           // sf row for A-fragment
    const unsigned nB = n0 + lane15;           // A-matrix column for B-fragment
    // A-frag (16-bit 16x32): lanes0-15 K0-15, lanes16-31 K16-31
    const __bf16* sfh = sf_hi + (size_t)mA * SFP + half * 16u;
    const __bf16* sfl = sf_lo + (size_t)mA * SFP + half * 16u;
    // B-frag (16-bit 32x16): lanes0-15 K0-15, lanes16-31 K16-31, col nB
    const __bf16* aph = Apk_hi + (size_t)nB * 32u + half * 16u;
    const __bf16* apl = Apk_lo + (size_t)nB * 32u + half * 16u;
    v8f acc = {};
    for (unsigned kc = 0; kc < NKC; ++kc) {
      const unsigned ko = kc * 32u;
      v16bf ah = ld16(sfh + ko, sfh + ko + 8u);
      v16bf al = ld16(sfl + ko, sfl + ko + 8u);
      const size_t bo = (size_t)kc * ((size_t)QP * 32u);
      v16bf bh = *(const v16bf*)(aph + bo);
      v16bf bl = *(const v16bf*)(apl + bo);
      // bf16x3: hi*hi + hi*lo + lo*hi, fp32 accumulate
      acc = __builtin_amdgcn_wmma_f32_16x16x32_bf16(false, ah, false, bh,
                                                    (short)0, acc, false, false);
      acc = __builtin_amdgcn_wmma_f32_16x16x32_bf16(false, ah, false, bl,
                                                    (short)0, acc, false, false);
      acc = __builtin_amdgcn_wmma_f32_16x16x32_bf16(false, al, false, bh,
                                                    (short)0, acc, false, false);
    }
    // epilogue: emission multiply, store U, deterministic half-wave row sums
    const unsigned q = n0 + lane15;
    const bool valid = q < QQ;
    float rowsum[8];
#pragma unroll
    for (int r = 0; r < 8; ++r) {
      const unsigned mrow = m0 + (unsigned)r + half * 8u;
      const float e = valid ? Et[(size_t)mrow * QP + q] : 0.f;
      const float v = acc[r] * e;
      U[(size_t)mrow * QP + q] = v;
      float s = v;
      s += __shfl_xor(s, 1);
      s += __shfl_xor(s, 2);
      s += __shfl_xor(s, 4);
      s += __shfl_xor(s, 8);
      rowsum[r] = s;
    }
    if (lane15 == 0u) {
#pragma unroll
      for (int r = 0; r < 8; ++r) {
        const unsigned mrow = m0 + (unsigned)r + half * 8u;
        Sp[mrow * NCT + ct] = rowsum[r];
      }
    }
  }
}

// Phase C: normalize, log output, ll update, bf16 hi/lo sf, next emission.
__device__ void phase_out(int t,
                          const float* __restrict__ x,
                          const float* __restrict__ Bm,
                          const float* __restrict__ U,
                          const float* __restrict__ Sp,
                          __bf16* __restrict__ sf_hi, __bf16* __restrict__ sf_lo,
                          float* __restrict__ Et,
                          float* __restrict__ ll0, float* __restrict__ ll1,
                          float* __restrict__ out, unsigned gtid) {
  const float* llr = (t & 1) ? ll0 : ll1;  // reads (t-1) parity buffer
  float* llw = (t & 1) ? ll1 : ll0;
  for (unsigned idx = gtid; idx < (unsigned)(BB * SFP); idx += NT) {
    const unsigned b = idx / SFP, q = idx % SFP;
    if (q >= QQ) {                          // keep K pads zero
      sf_hi[idx] = (__bf16)0.f;
      sf_lo[idx] = (__bf16)0.f;
      continue;
    }
    float S = 0.f;
    const float* sp = Sp + b * NCT;
#pragma unroll
    for (int c = 0; c < NCT; ++c) S += sp[c];  // fixed order: deterministic
    const float llold = (t == 0) ? 0.f : llr[b];
    const float lln = llold + logf(S);
    const unsigned uq = b * QP + q;
    const float sfv = U[uq] / S;
    const __bf16 h = (__bf16)sfv;
    sf_hi[idx] = h;
    sf_lo[idx] = (__bf16)(sfv - (float)h);
    out[(size_t)t * (BB * QQ) + (size_t)b * QQ + q] = logf(sfv + 1e-32f) + lln;
    if (q == 0u) llw[b] = lln;
    if (t + 1 < TT) {                        // E[t+1] for next step's GEMM
      const float* xr = x + ((size_t)(t + 1) * BB + b) * DD;
      Et[uq] = edot(xr, Bm + (size_t)q * DD);
    }
  }
}

__global__ __launch_bounds__(NTHR)
void hmm_forward_persistent(const float* __restrict__ x,
                            const float* __restrict__ Amat,
                            const float* __restrict__ Bm,
                            const float* __restrict__ pi,
                            float* __restrict__ out,
                            float* __restrict__ U, float* __restrict__ Et,
                            __bf16* __restrict__ sf_hi, __bf16* __restrict__ sf_lo,
                            __bf16* __restrict__ Apk_hi, __bf16* __restrict__ Apk_lo,
                            float* __restrict__ Sp,
                            float* __restrict__ ll0, float* __restrict__ ll1,
                            unsigned* __restrict__ counter) {
  const unsigned gtid = blockIdx.x * NTHR + threadIdx.x;
  const unsigned wave = gtid >> 5;
  const unsigned lane = gtid & 31u;
  unsigned phase = 0;

  // ---- Pack static A into B-fragment-native bf16 hi/lo layouts ----
  for (unsigned p = gtid; p < (unsigned)(NKC * QP); p += NT) {
    const unsigned kc = p / QP, n = p % QP;
    __bf16* dh = Apk_hi + (size_t)p * 32u;
    __bf16* dl = Apk_lo + (size_t)p * 32u;
#pragma unroll
    for (int k = 0; k < 32; ++k) {
      const unsigned kr = kc * 32u + (unsigned)k;
      const float a = (kr < QQ && n < QQ) ? Amat[(size_t)kr * QQ + n] : 0.f;
      const __bf16 h = (__bf16)a;
      dh[k] = h;
      dl[k] = (__bf16)(a - (float)h);
    }
  }

  // ---- Init (t=0): U = E0 * pi, deterministic per-(b,ct) partial sums ----
  for (unsigned idx = gtid; idx < (unsigned)(BB * NCT); idx += NT) {
    const unsigned b = idx / NCT, ct = idx % NCT;
    const float* xr = x + (size_t)b * DD;  // x[0,b,:]
    float s = 0.f;
#pragma unroll
    for (int j = 0; j < 16; ++j) {
      const unsigned q = ct * 16u + (unsigned)j;
      float u = 0.f;
      if (q < QQ) u = edot(xr, Bm + (size_t)q * DD) * pi[q];
      U[(size_t)b * QP + q] = u;
      s += u;
    }
    Sp[b * NCT + ct] = s;
  }
  grid_sync(counter, phase);
  phase_out(0, x, Bm, U, Sp, sf_hi, sf_lo, Et, ll0, ll1, out, gtid);
  grid_sync(counter, phase);

  // ---- Recurrence ----
  for (int t = 1; t < TT; ++t) {
    phase_gemm(Apk_hi, Apk_lo, sf_hi, sf_lo, Et, U, Sp, wave, lane);
    grid_sync(counter, phase);
    phase_out(t, x, Bm, U, Sp, sf_hi, sf_lo, Et, ll0, ll1, out, gtid);
    grid_sync(counter, phase);
  }
}

extern "C" void kernel_launch(void* const* d_in, const int* in_sizes, int n_in,
                              void* d_out, int out_size, void* d_ws, size_t ws_size,
                              hipStream_t stream) {
  const float* x  = (const float*)d_in[0];   // (T,B,D)
  const float* A  = (const float*)d_in[1];   // (Q,Q)
  const float* Bm = (const float*)d_in[2];   // (Q,D)
  const float* pi = (const float*)d_in[3];   // (Q,)
  float* out = (float*)d_out;                // (T,B,Q)

  // ws layout: [0,256) barrier counter; then payload (offsets stay 32B-aligned).
  char* base = (char*)d_ws;
  unsigned* counter = (unsigned*)base;
  float* ll0 = (float*)(base + 256);
  float* ll1 = ll0 + BB;
  float* Sp  = ll1 + BB;                     // BB*NCT
  float* U   = Sp + BB * NCT;                // BB*QP
  float* Et  = U + BB * QP;                  // BB*QP
  __bf16* sf_hi  = (__bf16*)(Et + BB * QP);  // BB*SFP
  __bf16* sf_lo  = sf_hi + (size_t)BB * SFP;
  __bf16* Apk_hi = sf_lo + (size_t)BB * SFP; // NKC*QP*32
  __bf16* Apk_lo = Apk_hi + (size_t)NKC * QP * 32;
  (void)in_sizes; (void)n_in; (void)out_size; (void)ws_size;

  hipMemsetAsync(d_ws, 0, 256, stream);  // reset grid barrier each call
  hipLaunchKernelGGL(hmm_forward_persistent, dim3(NBLK), dim3(NTHR), 0, stream,
                     x, A, Bm, pi, out, U, Et, sf_hi, sf_lo, Apk_hi, Apk_lo,
                     Sp, ll0, ll1, counter);
}
